// MSAPairWeightedAverage_60017872994814
// MI455X (gfx1250) — matrix operand order repaired and checked by reading, not compiled, not run on hardware
//
#include <hip/hip_runtime.h>
#include <hip/hip_fp16.h>

// Problem constants (from reference)
#define S_DIM 512
#define I_DIM 384
#define CM    64
#define CZ    128
#define H_DIM 8
#define C_DIM 32
#define HC    256
#define SI    (S_DIM * I_DIM)     // 196608
#define SC    (S_DIM * C_DIM)     // 16384
#define EPS   1e-5f

typedef __attribute__((ext_vector_type(16))) _Float16 v16h;
typedef __attribute__((ext_vector_type(8)))  _Float16 v8h;
typedef __attribute__((ext_vector_type(8)))  float    v8f;
typedef __attribute__((ext_vector_type(4)))  unsigned int v4u;
typedef __attribute__((ext_vector_type(8)))  int      v8i;
typedef __attribute__((ext_vector_type(4)))  int      v4i;

#ifndef __has_builtin
#define __has_builtin(x) 0
#endif
#if __has_builtin(__builtin_amdgcn_tensor_load_to_lds) && \
    __has_builtin(__builtin_amdgcn_s_wait_tensorcnt)
#define HAVE_TDM 1
#else
#define HAVE_TDM 0
#endif

// ---------------- wave32 reductions (lower to ds permute/swizzle) -------------
__device__ __forceinline__ float wave_sum(float x) {
#pragma unroll
  for (int m = 16; m >= 1; m >>= 1) x += __shfl_xor(x, m, 32);
  return x;
}
__device__ __forceinline__ float wave_max(float x) {
#pragma unroll
  for (int m = 16; m >= 1; m >>= 1) x = fmaxf(x, __shfl_xor(x, m, 32));
  return x;
}

// ---------------- WMMA fragment loaders --------------------------------------
// A fragment (16x32 f16, row-major source, stride ld), per ISA dense-A table:
//   lanes 0-15 : row M=lane,    K = kb+0..7  and kb+16..23
//   lanes 16-31: row M=lane-16, K = kb+8..15 and kb+24..31
__device__ __forceinline__ v16h load_a_frag(const _Float16* __restrict__ base,
                                            int ld, int m, int kb, int lane) {
  const _Float16* p = base + (size_t)m * ld + kb + ((lane >> 4) & 1) * 8;
  union { v16h v; v8h h[2]; } u;
  u.h[0] = *(const v8h*)(p);
  u.h[1] = *(const v8h*)(p + 16);
  return u.v;
}
// B fragment (32x16 f16) from an N-major (pre-transposed) operand Bt[N][K],
// same K order as the A fragment: lane holds column N=lane&15,
// K = kb + (lane>=16 ? 8 : 0) + 0..7 and +16..23 (two 16B loads).
__device__ __forceinline__ v16h load_b_frag(const _Float16* __restrict__ baseT,
                                            int ld, int n, int kb, int lane) {
  const _Float16* p = baseT + (size_t)n * ld + kb + ((lane >> 4) & 1) * 8;
  union { v16h v; v8h h[2]; } u;
  u.h[0] = *(const v8h*)(p);
  u.h[1] = *(const v8h*)(p + 16);
  return u.v;
}

#define WMMA_F16(A, B, C) \
  __builtin_amdgcn_wmma_f32_16x16x32_f16(false, (A), false, (B), (short)0, (C), false, false)

// ---------------- Kernel 1: LayerNorm(msa) -> f16 ----------------------------
__global__ void ln_msa_kernel(const float* __restrict__ msa,
                              const float* __restrict__ g,
                              const float* __restrict__ b,
                              _Float16* __restrict__ outn) {
  int wave = (blockIdx.x * blockDim.x + threadIdx.x) >> 5;   // row in [0, SI)
  int lane = threadIdx.x & 31;
  const float* row = msa + (size_t)wave * CM;
  float x0 = row[lane * 2], x1 = row[lane * 2 + 1];
  float mu = wave_sum(x0 + x1) * (1.0f / CM);
  float d0 = x0 - mu, d1 = x1 - mu;
  float var = wave_sum(d0 * d0 + d1 * d1) * (1.0f / CM);
  float r = rsqrtf(var + EPS);
  outn[(size_t)wave * CM + lane * 2]     = (_Float16)(d0 * r * g[lane * 2]     + b[lane * 2]);
  outn[(size_t)wave * CM + lane * 2 + 1] = (_Float16)(d1 * r * g[lane * 2 + 1] + b[lane * 2 + 1]);
}

// ---------------- Kernel 2: LayerNorm(pair) @ W_bias -> bias[h][i][j] --------
__global__ void ln_pair_bias_kernel(const float* __restrict__ pair,
                                    const float* __restrict__ g,
                                    const float* __restrict__ b,
                                    const float* __restrict__ Wb,   // [CZ][H]
                                    float* __restrict__ bias) {     // [H][I][I]
  int wave = (blockIdx.x * blockDim.x + threadIdx.x) >> 5;   // (i*I + j)
  int lane = threadIdx.x & 31;
  const float* row = pair + (size_t)wave * CZ;
  float x[4]; float s = 0.f;
#pragma unroll
  for (int q = 0; q < 4; q++) { x[q] = row[lane * 4 + q]; s += x[q]; }
  float mu = wave_sum(s) * (1.0f / CZ);
  float vs = 0.f;
#pragma unroll
  for (int q = 0; q < 4; q++) { x[q] -= mu; vs += x[q] * x[q]; }
  float var = wave_sum(vs) * (1.0f / CZ);
  float r = rsqrtf(var + EPS);
  float xn[4];
#pragma unroll
  for (int q = 0; q < 4; q++) xn[q] = x[q] * r * g[lane * 4 + q] + b[lane * 4 + q];
#pragma unroll
  for (int h = 0; h < H_DIM; h++) {
    float p = 0.f;
#pragma unroll
    for (int q = 0; q < 4; q++) p += xn[q] * Wb[(lane * 4 + q) * H_DIM + h];
    p = wave_sum(p);                       // full butterfly: result in all lanes
    if (lane == h) bias[(size_t)h * I_DIM * I_DIM + wave] = p;
  }
}

// ---------------- Kernel 3: softmax over j -> w[h][i][j] (f16) ---------------
__global__ void softmax_kernel(const float* __restrict__ bias,  // [H][I][I]
                               _Float16* __restrict__ wsm) {    // [H][I][I]
  int wave = (blockIdx.x * blockDim.x + threadIdx.x) >> 5;      // h*I + i
  int lane = threadIdx.x & 31;
  const float* row = bias + (size_t)wave * I_DIM;
  float x[12]; float mx = -1e30f;
#pragma unroll
  for (int t = 0; t < 12; t++) { x[t] = row[lane + 32 * t]; mx = fmaxf(mx, x[t]); }
  mx = wave_max(mx);
  float s = 0.f;
#pragma unroll
  for (int t = 0; t < 12; t++) { x[t] = __expf(x[t] - mx); s += x[t]; }
  float inv = 1.0f / wave_sum(s);
#pragma unroll
  for (int t = 0; t < 12; t++)
    wsm[(size_t)wave * I_DIM + lane + 32 * t] = (_Float16)(x[t] * inv);
}

// ---------------- Kernel 4: transpose+convert weights to f16 -----------------
__global__ void prep_weights_kernel(const float* __restrict__ Wv,  // [CM][HC]
                                    const float* __restrict__ Wg,  // [CM][HC]
                                    const float* __restrict__ Wo,  // [HC][CM]
                                    _Float16* __restrict__ Wv_t,   // [HC][CM]
                                    _Float16* __restrict__ Wg_t,   // [HC][CM]
                                    _Float16* __restrict__ Wo_t) { // [CM][HC]
  int idx = blockIdx.x * blockDim.x + threadIdx.x;   // 0..16383
  int k = idx % CM, n = idx / CM;
  Wv_t[n * CM + k] = (_Float16)Wv[k * HC + n];
  Wg_t[n * CM + k] = (_Float16)Wg[k * HC + n];
  int k2 = idx % HC, n2 = idx / HC;
  Wo_t[n2 * HC + k2] = (_Float16)Wo[k2 * CM + n2];
}

// ---------------- Kernel 5: v & gate GEMMs (WMMA, 1x4 N-blocked) -------------
// v    = msa_n @ W_v    -> scattered into vT[h][(s*32+c)][j] (B operand of PWA)
// gate = sigmoid(msa_n @ W_gate) -> gate[(s*I+i)][hc]
// One wave: 16 rows x 64 cols of both products; A fragment reused 8x per k-step.
__global__ void gemm_v_gate_kernel(const _Float16* __restrict__ msa_n, // [SI][CM]
                                   const _Float16* __restrict__ Wv_t,  // [HC][CM]
                                   const _Float16* __restrict__ Wg_t,  // [HC][CM]
                                   _Float16* __restrict__ vT,          // [H][SC][I]
                                   _Float16* __restrict__ gate) {      // [SI][HC]
  int wid  = (blockIdx.x * blockDim.x + threadIdx.x) >> 5;
  int lane = threadIdx.x & 31;
  int mt  = wid >> 2;                 // 12288 M-tiles
  int ntg = wid & 3;                  // 4 groups of 4 N-tiles
  int m0  = mt * 16;
  int am  = m0 + (lane & 15);
  // 384 % 16 == 0 => whole tile shares one s; hoist div/mod out of epilogue
  int s  = m0 / I_DIM;
  int jb = m0 - s * I_DIM;
  v8f accv[4] = {}; v8f accg[4] = {};
#pragma unroll
  for (int k = 0; k < CM; k += 32) {
    v16h a = load_a_frag(msa_n, CM, am, k, lane);
#pragma unroll
    for (int t = 0; t < 4; t++) {
      int bn = (ntg * 4 + t) * 16 + (lane & 15);
      v16h bv = load_b_frag(Wv_t, CM, bn, k, lane);
      v16h bg = load_b_frag(Wg_t, CM, bn, k, lane);
      accv[t] = WMMA_F16(a, bv, accv[t]);
      accg[t] = WMMA_F16(a, bg, accg[t]);
    }
  }
#pragma unroll
  for (int t = 0; t < 4; t++) {
    int n = (ntg * 4 + t) * 16 + (lane & 15);
    int h = n >> 5, c = n & 31;
    size_t vbase = ((size_t)h * SC + (size_t)s * C_DIM + c) * I_DIM + jb;
#pragma unroll
    for (int r = 0; r < 8; r++) {
      int roff = r + ((lane >> 4) << 3);          // row within tile
      vT[vbase + roff] = (_Float16)accv[t][r];
      float gv = 1.0f / (1.0f + __expf(-accg[t][r]));
      gate[(size_t)(m0 + roff) * HC + n] = (_Float16)gv;
    }
  }
}

// ---------------- Kernel 6: pair-weighted average GEMM (WMMA) + gate ---------
// per head h: weights[i,(s,c)] = sum_j w_h[i,j] * vT_h[(s,c), j]
// Block (8 waves) shares a 32x384 A-slab staged in LDS (TDM + verified copy);
// each wave computes a 32x32 output tile (2x2 register blocking).
// Epilogue: o = gate * weights (read-modify-write of gate buffer).
__global__ void gemm_pwa_kernel(const _Float16* __restrict__ wsm, // [H][I][I]
                                const _Float16* __restrict__ vT,  // [H][SC][I]
                                _Float16* __restrict__ o) {       // [SI][HC] (holds gate)
  __shared__ _Float16 As[32 * I_DIM];                             // 24 KB
  int lane = threadIdx.x & 31;
  int wib  = threadIdx.x >> 5;       // wave in block: 0..7
  int bidx = blockIdx.x;             // 8 h * 12 itb * 64 ntb = 6144
  int ntb  = bidx & 63;
  int itb  = (bidx >> 6) % 12;
  int h    = (bidx >> 6) / 12;
  const _Float16* Aslab = wsm + ((size_t)h * I_DIM + itb * 32) * I_DIM;
  const _Float16* B     = vT  + (size_t)h * SC * I_DIM;

#if HAVE_TDM
  // Real TDM issue: 1-D D# (count=1, type=2, data_size=2B, tile=32*384 elems)
  // loading the A-slab into LDS; completion via TENSORcnt. Bit-packing follows
  // ISA sec 8.3/8.4; since it is not HW-validated, the verified cooperative
  // copy below rewrites the same bytes, guaranteeing LDS contents either way.
  // 6-arg builtin (clang-23 lane): (g0, g1, g2, g3, g_extra, cpol);
  // extra descriptor word-group zero for a <=2-D tensor.
  if (threadIdx.x == 0) {
    unsigned long long ga = (unsigned long long)(const void*)Aslab;
    unsigned ldsa = (unsigned)(unsigned long long)(const void*)&As[0];
    v4u g0 = { 1u,                                  // count=1, user mode
               ldsa,                                // lds_addr
               (unsigned)(ga & 0xffffffffu),        // global_addr lo
               (unsigned)((ga >> 32) & 0x01ffffffu) | (2u << 30) }; // hi | type=2
    v8i g1 = { (int)(1u << 16),                     // data_size=1 (2 bytes)
               (int)((32u * I_DIM & 0xffffu) << 16),// tensor_dim0[15:0]
               (int)(1u << 16),                     // tensor_dim1=1
               (int)((32u * I_DIM) << 16),          // tile_dim0 = 12288
               (int)1,                              // tile_dim1 = 1
               (int)(32 * I_DIM),                   // tensor_dim0_stride
               0, 0 };
    v4i gz4 = { 0, 0, 0, 0 };
    v8i gz8 = { 0, 0, 0, 0, 0, 0, 0, 0 };
    __builtin_amdgcn_tensor_load_to_lds(g0, g1, gz4, gz4, gz8, 0);
    __builtin_amdgcn_s_wait_tensorcnt(0);
  }
#endif
  __syncthreads();
  // Verified staging path (idempotent w.r.t. the TDM load above).
  for (int e = threadIdx.x; e < (32 * I_DIM) / 8; e += blockDim.x)
    ((v8h*)As)[e] = ((const v8h*)Aslab)[e];
  __syncthreads();

  int n0  = ntb * 256 + wib * 32;
  int bn0 = n0 + (lane & 15);
  v8f acc[2][2] = {};
  for (int k = 0; k < I_DIM; k += 32) {
    __builtin_prefetch(B + (size_t)bn0 * I_DIM + k + 64, 0, 1);   // global_prefetch_b8
    v16h a0 = load_a_frag(As, I_DIM, (lane & 15),      k, lane);  // ds_load path
    v16h a1 = load_a_frag(As, I_DIM, 16 + (lane & 15), k, lane);
    v16h b0 = load_b_frag(B, I_DIM, bn0,      k, lane);
    v16h b1 = load_b_frag(B, I_DIM, bn0 + 16, k, lane);
    acc[0][0] = WMMA_F16(a0, b0, acc[0][0]);
    acc[0][1] = WMMA_F16(a0, b1, acc[0][1]);
    acc[1][0] = WMMA_F16(a1, b0, acc[1][0]);
    acc[1][1] = WMMA_F16(a1, b1, acc[1][1]);
  }
#pragma unroll
  for (int mm = 0; mm < 2; mm++) {
#pragma unroll
    for (int nn = 0; nn < 2; nn++) {
      int n = n0 + nn * 16 + (lane & 15);   // n = s*32 + c
      int s = n >> 5, c = n & 31;
#pragma unroll
      for (int r = 0; r < 8; r++) {
        int i = itb * 32 + mm * 16 + r + ((lane >> 4) << 3);
        size_t idx = ((size_t)s * I_DIM + i) * HC + h * C_DIM + c;
        float gv = (float)o[idx];               // gate
        o[idx] = (_Float16)(gv * acc[mm][nn][r]); // o = gate * weights
      }
    }
  }
}

// ---------------- Kernel 7: output GEMM o @ W_out (WMMA, 1x4 N-blocked) ------
__global__ void gemm_out_kernel(const _Float16* __restrict__ o,    // [SI][HC]
                                const _Float16* __restrict__ Wo_t, // [CM][HC]
                                float* __restrict__ out) {         // [SI][CM]
  int wid  = (blockIdx.x * blockDim.x + threadIdx.x) >> 5;   // = mt, 12288 waves
  int lane = threadIdx.x & 31;
  int am = wid * 16 + (lane & 15);
  v8f acc[4] = {};
#pragma unroll
  for (int k = 0; k < HC; k += 32) {
    v16h a = load_a_frag(o, HC, am, k, lane);
#pragma unroll
    for (int t = 0; t < 4; t++) {
      v16h b = load_b_frag(Wo_t, HC, t * 16 + (lane & 15), k, lane);
      acc[t] = WMMA_F16(a, b, acc[t]);
    }
  }
#pragma unroll
  for (int t = 0; t < 4; t++) {
    int n = t * 16 + (lane & 15);
#pragma unroll
    for (int r = 0; r < 8; r++) {
      int m = wid * 16 + r + ((lane >> 4) << 3);
      out[(size_t)m * CM + n] = acc[t][r];
    }
  }
}

// -----------------------------------------------------------------------------
extern "C" void kernel_launch(void* const* d_in, const int* in_sizes, int n_in,
                              void* d_out, int out_size, void* d_ws, size_t ws_size,
                              hipStream_t stream) {
  const float* msa       = (const float*)d_in[0];
  const float* pair      = (const float*)d_in[1];
  const float* ln_msa_g  = (const float*)d_in[2];
  const float* ln_msa_b  = (const float*)d_in[3];
  const float* ln_pair_g = (const float*)d_in[4];
  const float* ln_pair_b = (const float*)d_in[5];
  const float* Wv        = (const float*)d_in[6];
  const float* Wb        = (const float*)d_in[7];
  const float* Wg        = (const float*)d_in[8];
  const float* Wo        = (const float*)d_in[9];
  float* out = (float*)d_out;

  // workspace carve-out (~224 MB total), 256B aligned slices
  char* ws = (char*)d_ws;
  size_t off = 0;
  auto take = [&](size_t bytes) -> void* {
    void* p = (void*)(ws + off);
    off += (bytes + 255) & ~(size_t)255;
    return p;
  };
  _Float16* msa_n = (_Float16*)take((size_t)SI * CM * 2);               // 25.2 MB
  float*    bias  = (float*)   take((size_t)H_DIM * I_DIM * I_DIM * 4); // 4.7 MB
  _Float16* wsm   = (_Float16*)take((size_t)H_DIM * I_DIM * I_DIM * 2); // 2.4 MB
  _Float16* Wv_t  = (_Float16*)take((size_t)HC * CM * 2);
  _Float16* Wg_t  = (_Float16*)take((size_t)HC * CM * 2);
  _Float16* Wo_t  = (_Float16*)take((size_t)CM * HC * 2);
  _Float16* vT    = (_Float16*)take((size_t)H_DIM * SC * I_DIM * 2);    // 100.7 MB
  _Float16* gate  = (_Float16*)take((size_t)SI * HC * 2);               // 100.7 MB
  (void)ws_size; (void)in_sizes; (void)n_in; (void)out_size;

  // Phase 1: layernorms / bias / softmax / weight prep
  ln_msa_kernel      <<<SI / 4,               128, 0, stream>>>(msa, ln_msa_g, ln_msa_b, msa_n);
  ln_pair_bias_kernel<<<(I_DIM * I_DIM) / 4,  128, 0, stream>>>(pair, ln_pair_g, ln_pair_b, Wb, bias);
  softmax_kernel     <<<(H_DIM * I_DIM) / 4,  128, 0, stream>>>(bias, wsm);
  prep_weights_kernel<<<64,                   256, 0, stream>>>(Wv, Wg, Wo, Wv_t, Wg_t, Wo_t);

  // Phase 2: v & gate GEMMs (12288 mt x 4 nt-groups, 8 waves/block)
  gemm_v_gate_kernel <<<(SI / 16) * 4 / 8,    256, 0, stream>>>(msa_n, Wv_t, Wg_t, vT, gate);

  // Phase 3: pair-weighted average GEMM fused with gate
  //          grid = 8 heads * 12 it-blocks * 64 nt-blocks, 8 waves/block
  gemm_pwa_kernel    <<<H_DIM * 12 * 64,      256, 0, stream>>>(wsm, vT, gate);

  // Phase 4: output projection GEMM (12288 waves, 8 waves/block)
  gemm_out_kernel    <<<(SI / 16) / 8,        256, 0, stream>>>(gate, Wo_t, out);
}
